// QuantizedLinearWithlora_44899588112796
// MI455X (gfx1250) — compile-verified, matrix operand order checked by reading, not run on hardware
//
#include <hip/hip_runtime.h>
#include <hip/hip_bf16.h>

typedef __attribute__((ext_vector_type(16))) _Float16 v16h;
typedef __attribute__((ext_vector_type(8)))  _Float16 v8h;
typedef __attribute__((ext_vector_type(8)))  float    v8f;

#define NDIM 4096
#define SROWS 1024
#define LORA_ALPHA 0.8f
#define INV_CODEBOOK (1.0f / 1.03f)
#define FWHT_NORM 0.015625f   // 1/sqrt(4096)
#define BSTRIDE 72            // padded LDS row stride in halves (144 B = 36 banks,
                              // conflict-free across 16 consecutive rows; 16B-aligned)

// ---------------------------------------------------------------------------
// K1: row-wise normalized FWHT of (src * SU), fp32 -> fp16 output.
// ---------------------------------------------------------------------------
__global__ __launch_bounds__(256)
void qlora_fwht_rows_kernel(const float* __restrict__ src,
                            const float* __restrict__ su,
                            _Float16* __restrict__ dst) {
  __shared__ float buf[NDIM];
  const int row = blockIdx.x;
  const float* s = src + (size_t)row * NDIM;
  for (int i = threadIdx.x; i < NDIM; i += 256)
    buf[i] = s[i] * su[i];
  for (int h = 1; h < NDIM; h <<= 1) {
    __syncthreads();
    for (int t = threadIdx.x; t < NDIM / 2; t += 256) {
      int idx = ((t & ~(h - 1)) << 1) | (t & (h - 1));
      float a = buf[idx], b = buf[idx + h];
      buf[idx]     = a + b;
      buf[idx + h] = a - b;
    }
  }
  __syncthreads();
  _Float16* d = dst + (size_t)row * NDIM;
  for (int i = threadIdx.x; i < NDIM; i += 256)
    d[i] = (_Float16)(buf[i] * FWHT_NORM);
}

// ---------------------------------------------------------------------------
// K2: E8P decode. One 16-bit codeword -> 8 fp16 weights (exact multiples of
// 0.25; the 1/1.03 codebook scale is folded into the output epilogue).
// ---------------------------------------------------------------------------
__global__ __launch_bounds__(256)
void qlora_decode_e8p_kernel(const int* __restrict__ codes,
                             const int* __restrict__ grid,
                             _Float16* __restrict__ Wh,
                             int ncodes) {
  __shared__ unsigned sgrid[256];
  const int tid = threadIdx.x;
  sgrid[tid] = (unsigned)grid[tid];
  __syncthreads();
  const int i = blockIdx.x * 256 + tid;
  if (i >= ncodes) return;
  const int c = codes[i];
  const unsigned packed = sgrid[c & 255];
  const int sbits = (c >> 8) & 127;
  const float delta = ((c >> 15) & 1) ? 0.25f : -0.25f;
  const float s8 = (__popc(sbits) & 1) ? -1.0f : 1.0f;
  v8h w;
#pragma unroll
  for (int j = 0; j < 8; ++j) {
    float nib = (float)((packed >> (4 * j)) & 15u);
    float sg = (j < 7) ? (((sbits >> j) & 1) ? -1.0f : 1.0f) : s8;
    w[j] = (_Float16)(sg * 0.5f * nib + delta);
  }
  *(v8h*)(Wh + (size_t)i * 8) = w;
}

// ---------------------------------------------------------------------------
// K3: WMMA GEMM  Y[s][m] = sum_n A[s][n] * B[m][n]   (A,B fp16; Y fp32)
// WG = 256 threads = 8 waves; each wave owns a 16(s) x 64(m) output block,
// block covers 128 s-rows x 64 m-cols. B tiles (64m x 64k halves, row-padded
// to 72 halves against LDS bank conflicts) are staged in a DOUBLE-BUFFERED
// LDS region using CDNA5 async global->LDS DMA (ASYNCcnt). One barrier per
// 64-wide K-step, 8 v_wmma per barrier:
//   iter i: issue 2x global_load_async_to_lds_b128 tile i+1 -> buf[(i+1)&1]
//           8x v_wmma on tile i from buf[i&1]
//           s_wait_asynccnt 0 ; s_barrier
// Buffer (i+1)&1's previous readers (tile i-1) finished before the barrier
// at the end of iter i-1, so the async overwrite is race-free.
// ---------------------------------------------------------------------------
__global__ __launch_bounds__(256)
void qlora_gemm_wmma_kernel(const _Float16* __restrict__ A,   // [S][4096]
                            const _Float16* __restrict__ B,   // [Mout][4096]
                            float* __restrict__ Y,            // [S][Mout]
                            int Mout) {
  __shared__ _Float16 bt[2][64 * BSTRIDE];
  const int tid  = threadIdx.x;
  const int lane = tid & 31;
  const int wave = tid >> 5;
  const int m0 = blockIdx.x * 64;
  const int s0 = blockIdx.y * 128 + wave * 16;
  const int off  = (lane >> 4) * 8;   // lanes 0-15: K 0..7,16..23 ; 16-31: K 8..15,24..31
  const int rsel = lane & 15;

  const _Float16* arow = A + (size_t)(s0 + rsel) * NDIM;
  const int bm = tid >> 2;            // 0..63  (m within tile)
  const int bc = (tid & 3) * 16;      // 0,16,32,48 (k chunk: 32 B per thread)
  const _Float16* gsrc = B + (size_t)(m0 + bm) * NDIM + bc;

  // LDS byte offsets for the async DMA destination (low 32 bits of the
  // generic shared pointer are the LDS offset). The instruction's imm offset
  // is added to BOTH the LDS and global addresses (ISA 10.x pseudocode), so
  // one address pair covers both 16 B chunks via offset:0 / offset:16.
  unsigned lds0 = (unsigned)(size_t)&bt[0][bm * BSTRIDE + bc];
  unsigned lds1 = (unsigned)(size_t)&bt[1][bm * BSTRIDE + bc];

  // Prologue: async-copy tile 0 into buffer 0 (32 B per thread).
  asm volatile("global_load_async_to_lds_b128 %0, %1, off"
               :: "v"(lds0), "v"((unsigned long long)(size_t)gsrc)
               : "memory");
  asm volatile("global_load_async_to_lds_b128 %0, %1, off offset:16"
               :: "v"(lds0), "v"((unsigned long long)(size_t)gsrc)
               : "memory");
  asm volatile("s_wait_asynccnt 0x0" ::: "memory");
  __syncthreads();

  v8f acc[4] = {};

  for (int i = 0; i < NDIM / 64; ++i) {
    const int cur = i & 1;
    if (i + 1 < NDIM / 64) {
      const _Float16* g = gsrc + (size_t)(i + 1) * 64;
      unsigned ldst = cur ? lds0 : lds1;
      asm volatile("global_load_async_to_lds_b128 %0, %1, off"
                   :: "v"(ldst), "v"((unsigned long long)(size_t)g)
                   : "memory");
      asm volatile("global_load_async_to_lds_b128 %0, %1, off offset:16"
                   :: "v"(ldst), "v"((unsigned long long)(size_t)g)
                   : "memory");
    }
    const int k0 = i * 64;
    const _Float16* bbase = &bt[cur][0];
#pragma unroll
    for (int j = 0; j < 2; ++j) {           // two 32-wide K sub-steps
      v8h alo = *(const v8h*)(arow + k0 + j * 32 + off);
      v8h ahi = *(const v8h*)(arow + k0 + j * 32 + off + 16);
      v16h a = __builtin_shufflevector(alo, ahi, 0, 1, 2, 3, 4, 5, 6, 7,
                                       8, 9, 10, 11, 12, 13, 14, 15);
#pragma unroll
      for (int t = 0; t < 4; ++t) {
        const _Float16* bp = bbase + (t * 16 + rsel) * BSTRIDE + j * 32 + off;
        v8h blo = *(const v8h*)bp;
        v8h bhi = *(const v8h*)(bp + 16);
        v16h b = __builtin_shufflevector(blo, bhi, 0, 1, 2, 3, 4, 5, 6, 7,
                                         8, 9, 10, 11, 12, 13, 14, 15);
        acc[t] = __builtin_amdgcn_wmma_f32_16x16x32_f16(
            false, a, false, b, (short)0, acc[t], false, false);
      }
    }
    asm volatile("s_wait_asynccnt 0x0" ::: "memory");
    __syncthreads();
  }

  // C/D layout: element D[M=r + 8*(lane>=16)][N=lane%16] in VGPR r.
  const int srow = s0 + (lane >> 4) * 8;
#pragma unroll
  for (int t = 0; t < 4; ++t) {
    const int mcol = m0 + t * 16 + rsel;
#pragma unroll
    for (int r = 0; r < 8; ++r)
      Y[(size_t)(srow + r) * Mout + mcol] = acc[t][r];
  }
}

// ---------------------------------------------------------------------------
// K4: output FWHT over m, scale by SV*Wscale/(1.03*64), add rank-64 LoRA up-
// projection (lr row cached in LDS, lora_A streamed from L2 as float4).
// ---------------------------------------------------------------------------
__global__ __launch_bounds__(256)
void qlora_fwht_out_kernel(const float* __restrict__ y,      // [S][4096]
                           const float* __restrict__ lr,     // [S][64]
                           const float* __restrict__ loraA,  // [4096][64]
                           const float* __restrict__ sv,
                           const float* __restrict__ wscale_p,
                           float* __restrict__ out) {
  __shared__ float buf[NDIM];
  __shared__ float lrs[64];
  const int row = blockIdx.x;
  const float* ysrc = y + (size_t)row * NDIM;
  for (int i = threadIdx.x; i < NDIM; i += 256)
    buf[i] = ysrc[i];
  if (threadIdx.x < 64)
    lrs[threadIdx.x] = lr[row * 64 + threadIdx.x];
  for (int h = 1; h < NDIM; h <<= 1) {
    __syncthreads();
    for (int t = threadIdx.x; t < NDIM / 2; t += 256) {
      int idx = ((t & ~(h - 1)) << 1) | (t & (h - 1));
      float a = buf[idx], b = buf[idx + h];
      buf[idx]     = a + b;
      buf[idx + h] = a - b;
    }
  }
  __syncthreads();
  const float ws = wscale_p[0] * INV_CODEBOOK * FWHT_NORM;
  float* orow = out + (size_t)row * NDIM;
  for (int m = threadIdx.x; m < NDIM; m += 256) {
    const float4* ap = (const float4*)(loraA + (size_t)m * 64);
    float acc = 0.0f;
#pragma unroll
    for (int r4 = 0; r4 < 16; ++r4) {
      float4 a4 = ap[r4];
      float4 l4 = ((const float4*)lrs)[r4];
      acc += a4.x * l4.x + a4.y * l4.y + a4.z * l4.z + a4.w * l4.w;
    }
    orow[m] = buf[m] * ws * sv[m] + LORA_ALPHA * acc;
  }
}

// ---------------------------------------------------------------------------
// Launch: z = fwht(x*SU); loraBt = fwht(lora_B*SU); Wh = decode(Qidxs);
// y = z*Wh^T (WMMA); lr = z*loraBt^T (WMMA); out = fwht(y)*scale + a*lr*A^T.
// ---------------------------------------------------------------------------
extern "C" void kernel_launch(void* const* d_in, const int* in_sizes, int n_in,
                              void* d_out, int out_size, void* d_ws, size_t ws_size,
                              hipStream_t stream) {
  const float* x      = (const float*)d_in[0];   // [1,1024,4096]
  const float* SU     = (const float*)d_in[1];   // [4096]
  const float* SV     = (const float*)d_in[2];   // [4096]
  const float* Wscale = (const float*)d_in[3];   // [1]
  const float* lora_A = (const float*)d_in[4];   // [4096,64]
  const float* lora_B = (const float*)d_in[5];   // [64,4096]
  const int*   Qidxs  = (const int*)d_in[6];     // [4096,512]
  const int*   grid   = (const int*)d_in[7];     // [256]
  float* out = (float*)d_out;

  char* ws = (char*)d_ws;
  _Float16* z      = (_Float16*)(ws);                     //  8 MB: [1024][4096] f16
  _Float16* loraBt = (_Float16*)(ws + 8388608);           // 512 KB: [64][4096] f16
  _Float16* Wh     = (_Float16*)(ws + 8912896);           // 32 MB: [4096][4096] f16
  float*    y      = (float*)   (ws + 42467328);          // 16 MB: [1024][4096] f32
  float*    lr     = (float*)   (ws + 59244544);          // 256 KB: [1024][64] f32

  // Input transform (z) and LoRA-B transform (reuses the same kernel).
  qlora_fwht_rows_kernel<<<SROWS, 256, 0, stream>>>(x, SU, z);
  qlora_fwht_rows_kernel<<<64, 256, 0, stream>>>(lora_B, SU, loraBt);

  // Dequantize E8P codebook weights to fp16 (exact; 1/1.03 deferred).
  const int ncodes = 4096 * 512;
  qlora_decode_e8p_kernel<<<ncodes / 256, 256, 0, stream>>>(Qidxs, grid, Wh, ncodes);

  // Main GEMM: 1024x4096x4096, and LoRA-down GEMM: 1024x64x4096.
  qlora_gemm_wmma_kernel<<<dim3(64, 8), 256, 0, stream>>>(z, Wh, y, 4096);
  qlora_gemm_wmma_kernel<<<dim3(1, 8), 256, 0, stream>>>(z, loraBt, lr, 64);

  // Output transform + scaling + rank-64 LoRA up-projection.
  qlora_fwht_out_kernel<<<SROWS, 256, 0, stream>>>(y, lr, lora_A, SV, Wscale, out);
}
